// LSTM_76321568849957
// MI455X (gfx1250) — matrix operand order repaired and checked by reading, not compiled, not run on hardware
//
#include <hip/hip_runtime.h>
#include <hip/hip_bf16.h>

// ---------------------------------------------------------------------------
// LSTM on gfx1250: per-timestep kernel, 4-gate bf16 WMMA + f32 cell update,
// double-buffered LDS pipeline with async global->LDS weight staging.
// ---------------------------------------------------------------------------

typedef __attribute__((ext_vector_type(16))) __bf16 v16bf;
typedef __attribute__((ext_vector_type(8)))  float  v8f;
typedef int v4i_ls __attribute__((vector_size(4 * sizeof(int))));

#define HID   512
#define BAT   512
#define SEQ   512
#define IND   64
#define NCLS  10

#if defined(__has_builtin)
#if __has_builtin(__builtin_amdgcn_global_load_async_to_lds_b128)
#define HAVE_ASYNC_LDS 1
#endif
#endif
#ifndef HAVE_ASYNC_LDS
#define HAVE_ASYNC_LDS 0
#endif

#define GLOBAL_AS __attribute__((address_space(1)))
#define LDS_AS    __attribute__((address_space(3)))

__device__ __forceinline__ unsigned short f2bf(float f) {
    unsigned u = __builtin_bit_cast(unsigned, f);
    unsigned r = u + 0x7FFFu + ((u >> 16) & 1u);   // round-to-nearest-even
    return (unsigned short)(r >> 16);
}

__device__ __forceinline__ float sigm(float x) {
    return 1.0f / (1.0f + expf(-x));
}

union FragBF {
    v16bf v;
    uint4 q[2];
};

// --------------------------- helper kernels --------------------------------

__global__ void zero_f32(float* __restrict__ p, int n) {
    int i = blockIdx.x * blockDim.x + threadIdx.x;
    if (i < n) p[i] = 0.0f;
}

__global__ void cvt_bf16(const float* __restrict__ in, unsigned short* __restrict__ out, int n) {
    int i = blockIdx.x * blockDim.x + threadIdx.x;
    if (i < n) out[i] = f2bf(in[i]);
}

// --------------------------- per-step kernel -------------------------------
// Block: 128 threads (4 waves). Block computes a 32x32 (hidden x batch) tile
// of all 4 gate pre-activations via WMMA, then does the LSTM cell update.
// K loop: 16 chunks of 32 over the recurrence (HID) + 2 chunks over the
// input projection (IND), software-pipelined through 2 LDS buffers.

__global__ __launch_bounds__(128)
void lstm_step(const unsigned short* __restrict__ Whb,   // [4][HID][HID] bf16 (g,i,f,o)
               const unsigned short* __restrict__ Wxb,   // [4][HID][IND] bf16
               const float* __restrict__ bg, const float* __restrict__ bi,
               const float* __restrict__ bfv, const float* __restrict__ bo,
               const float* __restrict__ x,  int t,
               const float* __restrict__ hin, float* __restrict__ hout,
               float* __restrict__ cst)
{
    __shared__ __align__(16) unsigned short sW[2][4][32][32];   // [buf][gate][m][k]
    __shared__ __align__(16) unsigned short sBT[2][32][32];     // [buf][n][k] (transposed)

    const int tid  = threadIdx.x;
    const int lane = tid & 31;
    const int w    = tid >> 5;
    const int mt   = w >> 1;          // 0..1 : which 16-row sub-tile
    const int nt   = w & 1;           // 0..1 : which 16-col sub-tile

    const int N0 = blockIdx.x * 32;   // batch base
    const int M0 = blockIdx.y * 32;   // hidden-row base

    const int half = lane >> 4;       // 0: lanes 0-15, 1: lanes 16-31
    const int l15  = lane & 15;

    v8f acc[4] = {};

    const int NCHUNK = (HID / 32) + (IND / 32);   // 16 + 2 = 18

    // stage K-chunk `it` into LDS buffer `buf`
    auto stage = [&](int it, int buf) {
        const bool isH = (it < HID / 32);
        const int  k0  = isH ? (it << 5) : ((it - HID / 32) << 5);
        // ---- weight tiles: 4 gates x 32 rows x 32 k of bf16, 16B per thread-chunk
        #pragma unroll
        for (int cch = 0; cch < 4; ++cch) {
            int cidx = tid + cch * 128;          // 0..511
            int g    = cidx >> 7;
            int rem  = cidx & 127;
            int m    = rem >> 2;
            int k8   = (rem & 3) << 3;
            const unsigned short* src = isH
                ? Whb + (size_t)g * HID * HID + (size_t)(M0 + m) * HID + k0 + k8
                : Wxb + (size_t)g * HID * IND + (size_t)(M0 + m) * IND + k0 + k8;
            unsigned short* dst = &sW[buf][g][m][k8];
#if HAVE_ASYNC_LDS
            __builtin_amdgcn_global_load_async_to_lds_b128(
                (GLOBAL_AS v4i_ls*)(uintptr_t)src,
                (LDS_AS v4i_ls*)(uintptr_t)dst, 0, 0);
#else
            *reinterpret_cast<uint4*>(dst) = *reinterpret_cast<const uint4*>(src);
#endif
        }
        // ---- B tile (f32 -> bf16), stored transposed [n][k]
        #pragma unroll
        for (int j = 0; j < 2; ++j) {
            int ch = tid + j * 128;              // 0..255
            if (isH) {
                int r  = ch >> 3;                // k within chunk
                int n4 = (ch & 7) << 2;
                const float4 v = *reinterpret_cast<const float4*>(
                    hin + (size_t)(k0 + r) * BAT + N0 + n4);
                sBT[buf][n4 + 0][r] = f2bf(v.x);
                sBT[buf][n4 + 1][r] = f2bf(v.y);
                sBT[buf][n4 + 2][r] = f2bf(v.z);
                sBT[buf][n4 + 3][r] = f2bf(v.w);
            } else {
                int n  = ch >> 3;
                int r4 = (ch & 7) << 2;
                const float4 v = *reinterpret_cast<const float4*>(
                    x + ((size_t)(N0 + n) * SEQ + t) * IND + k0 + r4);
                sBT[buf][n][r4 + 0] = f2bf(v.x);
                sBT[buf][n][r4 + 1] = f2bf(v.y);
                sBT[buf][n][r4 + 2] = f2bf(v.z);
                sBT[buf][n][r4 + 3] = f2bf(v.w);
            }
        }
    };

    // prologue: stage chunk 0
    stage(0, 0);
#if HAVE_ASYNC_LDS
    asm volatile("s_wait_asynccnt 0x0" ::: "memory");
#endif
    __syncthreads();

    for (int it = 0; it < NCHUNK; ++it) {
        const int cur = it & 1;
        if (it + 1 < NCHUNK) stage(it + 1, cur ^ 1);

        // B fragment: k = half*16 + e, n = nt*16 + l15  (two ds_load_b128)
        FragBF fb;
        {
            const uint4* p = reinterpret_cast<const uint4*>(
                &sBT[cur][nt * 16 + l15][half * 16]);
            fb.q[0] = p[0];
            fb.q[1] = p[1];
        }
        // A fragments per gate: m = mt*16 + l15, k = half*16 + e
        #pragma unroll
        for (int g = 0; g < 4; ++g) {
            FragBF fa;
            fa.q[0] = *reinterpret_cast<const uint4*>(
                &sW[cur][g][mt * 16 + l15][half * 16]);
            fa.q[1] = *reinterpret_cast<const uint4*>(
                &sW[cur][g][mt * 16 + l15][half * 16 + 8]);
            acc[g] = __builtin_amdgcn_wmma_f32_16x16x32_bf16(
                false, fa.v, false, fb.v, (short)0, acc[g], false, false);
        }

#if HAVE_ASYNC_LDS
        asm volatile("s_wait_asynccnt 0x0" ::: "memory");
#endif
        __syncthreads();
    }

    // ---------------- LSTM cell update for this wave's 16x16 tile ----------
    // D layout: VGPR r -> row = half*8 + r, col = l15
    const int gcol = N0 + nt * 16 + l15;
    #pragma unroll
    for (int r = 0; r < 8; ++r) {
        const int grow = M0 + mt * 16 + (half << 3) + r;
        const size_t idx = (size_t)grow * BAT + gcol;
        const float pg = tanhf(acc[0][r] + bg[grow]);
        const float pi = sigm(acc[1][r] + bi[grow]);
        const float pf = sigm(acc[2][r] + bfv[grow]);
        const float po = sigm(acc[3][r] + bo[grow]);
        const float cold = cst[idx];
        const float cn = pg * pi + cold * pf;
        cst[idx]  = cn;
        hout[idx] = tanhf(cn) * po;
    }
}

// --------------------------- final projection ------------------------------

__global__ __launch_bounds__(128)
void proj_kernel(const float* __restrict__ h, const float* __restrict__ Wph,
                 const float* __restrict__ bp, float* __restrict__ out)
{
    int b = blockIdx.x * blockDim.x + threadIdx.x;   // batch index
    if (b >= BAT) return;
    float acc[NCLS];
    #pragma unroll
    for (int c = 0; c < NCLS; ++c) acc[c] = 0.0f;
    for (int hid = 0; hid < HID; ++hid) {
        float hv = h[(size_t)hid * BAT + b];
        #pragma unroll
        for (int c = 0; c < NCLS; ++c)
            acc[c] += Wph[(size_t)c * HID + hid] * hv;
    }
    #pragma unroll
    for (int c = 0; c < NCLS; ++c)
        out[(size_t)b * NCLS + c] = acc[c] + bp[c];
}

// ---------------------------------------------------------------------------

extern "C" void kernel_launch(void* const* d_in, const int* in_sizes, int n_in,
                              void* d_out, int out_size, void* d_ws, size_t ws_size,
                              hipStream_t stream) {
    (void)in_sizes; (void)n_in; (void)out_size; (void)ws_size;

    const float* x   = (const float*)d_in[0];
    const float* Wgx = (const float*)d_in[1];
    const float* Wgh = (const float*)d_in[2];
    const float* bg  = (const float*)d_in[3];
    const float* Wix = (const float*)d_in[4];
    const float* Wih = (const float*)d_in[5];
    const float* bi  = (const float*)d_in[6];
    const float* Wfx = (const float*)d_in[7];
    const float* Wfh = (const float*)d_in[8];
    const float* bfv = (const float*)d_in[9];
    const float* Wox = (const float*)d_in[10];
    const float* Woh = (const float*)d_in[11];
    const float* bo  = (const float*)d_in[12];
    const float* Wph = (const float*)d_in[13];
    const float* bp  = (const float*)d_in[14];
    float* out = (float*)d_out;

    // workspace layout
    char* w = (char*)d_ws;
    float*          hA  = (float*)(w + 0);                       // 1 MB
    float*          hB  = (float*)(w + (1u << 20));              // 1 MB
    float*          cS  = (float*)(w + (2u << 20));              // 1 MB
    unsigned short* Whb = (unsigned short*)(w + (3u << 20));     // 2 MB
    unsigned short* Wxb = (unsigned short*)(w + (5u << 20));     // 256 KB

    // zero initial h and c
    zero_f32<<<(HID * BAT + 255) / 256, 256, 0, stream>>>(hA, HID * BAT);
    zero_f32<<<(HID * BAT + 255) / 256, 256, 0, stream>>>(cS, HID * BAT);

    // convert weights to bf16 once: gate order g,i,f,o
    const int nh = HID * HID, nx = HID * IND;
    cvt_bf16<<<(nh + 255) / 256, 256, 0, stream>>>(Wgh, Whb + 0 * (size_t)nh, nh);
    cvt_bf16<<<(nh + 255) / 256, 256, 0, stream>>>(Wih, Whb + 1 * (size_t)nh, nh);
    cvt_bf16<<<(nh + 255) / 256, 256, 0, stream>>>(Wfh, Whb + 2 * (size_t)nh, nh);
    cvt_bf16<<<(nh + 255) / 256, 256, 0, stream>>>(Woh, Whb + 3 * (size_t)nh, nh);
    cvt_bf16<<<(nx + 255) / 256, 256, 0, stream>>>(Wgx, Wxb + 0 * (size_t)nx, nx);
    cvt_bf16<<<(nx + 255) / 256, 256, 0, stream>>>(Wix, Wxb + 1 * (size_t)nx, nx);
    cvt_bf16<<<(nx + 255) / 256, 256, 0, stream>>>(Wfx, Wxb + 2 * (size_t)nx, nx);
    cvt_bf16<<<(nx + 255) / 256, 256, 0, stream>>>(Wox, Wxb + 3 * (size_t)nx, nx);

    // recurrence: one kernel per timestep, double-buffered h
    dim3 grid(BAT / 32, HID / 32);
    for (int t = 0; t < SEQ; ++t) {
        const float* hi = (t & 1) ? hB : hA;
        float*       ho = (t & 1) ? hA : hB;
        lstm_step<<<grid, 128, 0, stream>>>(Whb, Wxb, bg, bi, bfv, bo,
                                            x, t, hi, ho, cS);
    }
    // SEQ = 512 (even): final h is in hA
    proj_kernel<<<(BAT + 127) / 128, 128, 0, stream>>>(hA, Wph, bp, out);
}
